// UrbanODE_87428354278231
// MI455X (gfx1250) — matrix operand order repaired and checked by reading, not compiled, not run on hardware
//
#include <hip/hip_runtime.h>
#include <math.h>
#include <stdint.h>

// ---------------- problem constants ----------------
#define BATCH 32
#define CH    64
#define CH2   32
#define HW    1024          // 32x32
#define NTOT  3278          // 1024+784+625+484+361
#define NP    3328          // padded to multiple of 128 (26*128)
#define KQ    288           // C2*9
#define KQP   320           // KQ padded to multiple of 64 (5*64)
#define KR    576           // C*9
#define KRP   640           // KR padded to multiple of 128 (5*128)

typedef _Float16 h8   __attribute__((ext_vector_type(8)));
typedef _Float16 h16  __attribute__((ext_vector_type(16)));
typedef float    f8   __attribute__((ext_vector_type(8)));

// decode global patch index -> (grid size hs, local pixel p)
__device__ __forceinline__ void decode_n(int n, int& hs, int& p){
  if (n < 1024)      { hs = 32; p = n;        }
  else if (n < 1808) { hs = 28; p = n - 1024; }
  else if (n < 2433) { hs = 25; p = n - 1808; }
  else if (n < 2917) { hs = 22; p = n - 2433; }
  else               { hs = 19; p = n - 2917; }
}

// ---------------- kernel 1: 1x1 convs + PReLU (full-res + pyramid) ----------------
__global__ __launch_bounds__(256)
void act_kernel(const float* __restrict__ xin,
                const float* __restrict__ wb, const float* __restrict__ bb, const float* __restrict__ ab,
                const float* __restrict__ wm, const float* __restrict__ bm, const float* __restrict__ am,
                const float* __restrict__ wa, const float* __restrict__ ba, const float* __restrict__ aa,
                float* __restrict__ base, float* __restrict__ refm, float* __restrict__ mb)
{
  const int T1 = BATCH * CH  * NTOT;
  const int T2 = T1 + BATCH * CH2 * NTOT;
  const int T3 = T2 + BATCH * CH2 * HW;
  for (int idx = blockIdx.x * blockDim.x + threadIdx.x; idx < T3;
       idx += gridDim.x * blockDim.x) {
    if (idx < T1) {
      int n = idx % NTOT; int c = (idx / NTOT) % CH; int b = idx / (NTOT * CH);
      int hs, p; decode_n(n, hs, p);
      int y = p / hs, x = p % hs;
      int iy = (y * 32) / hs, ix = (x * 32) / hs;       // nearest, floor(dst*in/out)
      const float* px = xin + (size_t)b * CH * HW + iy * 32 + ix;
      float s = ba[c];
      #pragma unroll 8
      for (int ci = 0; ci < CH; ci++) s += wa[c * CH + ci] * px[(size_t)ci * HW];
      base[idx] = s >= 0.f ? s : aa[0] * s;
    } else if (idx < T2) {
      int j = idx - T1;
      int n = j % NTOT; int c = (j / NTOT) % CH2; int b = j / (NTOT * CH2);
      int hs, p; decode_n(n, hs, p);
      int y = p / hs, x = p % hs;
      int iy = (y * 32) / hs, ix = (x * 32) / hs;
      const float* px = xin + (size_t)b * CH * HW + iy * 32 + ix;
      float s = bm[c];
      #pragma unroll 8
      for (int ci = 0; ci < CH; ci++) s += wm[c * CH + ci] * px[(size_t)ci * HW];
      refm[j] = s >= 0.f ? s : am[0] * s;
    } else {
      int j = idx - T2;
      int p = j % HW; int c = (j / HW) % CH2; int b = j / (HW * CH2);
      const float* px = xin + (size_t)b * CH * HW + p;
      float s = bb[c];
      #pragma unroll 8
      for (int ci = 0; ci < CH; ci++) s += wb[c * CH + ci] * px[(size_t)ci * HW];
      mb[j] = s >= 0.f ? s : ab[0] * s;
    }
  }
}

// ---------------- kernel 2: query patches xp [B][HW][KQP] (f16, K zero-padded) ----------------
__global__ __launch_bounds__(256)
void xp_kernel(const float* __restrict__ mb, _Float16* __restrict__ xp)
{
  const int TOT = BATCH * HW * KQP;
  for (int idx = blockIdx.x * blockDim.x + threadIdx.x; idx < TOT;
       idx += gridDim.x * blockDim.x) {
    int k = idx % KQP; int p = (idx / KQP) % HW; int b = idx / (KQP * HW);
    float v = 0.f;
    if (k < KQ) {
      int c = k / 9, r9 = k % 9, dy = r9 / 3, dx = r9 % 3;
      int y = p >> 5, x = p & 31;
      int yy = y + dy - 1, xx = x + dx - 1;
      if (yy >= 0 && yy < 32 && xx >= 0 && xx < 32)
        v = mb[((size_t)b * CH2 + c) * HW + yy * 32 + xx];
    }
    xp[idx] = (_Float16)v;
  }
}

// ---------------- kernel 3: filter patches + L2 norm -> wi [B][NP][KQP] (f16) ----------------
__global__ __launch_bounds__(256)
void wi_kernel(const float* __restrict__ refm, _Float16* __restrict__ wi)
{
  int wave = (int)((blockIdx.x * blockDim.x + threadIdx.x) >> 5);
  int lane = threadIdx.x & 31;
  const int total = BATCH * NP;
  if (wave >= total) return;                       // wave-uniform
  int n = wave % NP; int b = wave / NP;
  _Float16* W = wi + ((size_t)b * NP + n) * KQP;
  if (n >= NTOT) {                                 // M-padding rows: zero
    for (int k = lane; k < KQP; k += 32) W[k] = (_Float16)0.f;
    return;
  }
  int hs, p; decode_n(n, hs, p);
  int y = p / hs, x = p % hs;
  float vals[10];
  float ss = 0.f;
  #pragma unroll
  for (int i = 0; i < 10; i++) {
    int k = lane + i * 32;
    float v = 0.f;
    if (k < KQ) {
      int c = k / 9, r9 = k % 9, dy = r9 / 3, dx = r9 % 3;
      int yy = y + dy - 1, xx = x + dx - 1;
      if (yy >= 0 && yy < hs && xx >= 0 && xx < hs) {
        int n2 = n + (dy - 1) * hs + (dx - 1);     // neighbor inside same pyramid level
        v = refm[((size_t)b * CH2 + c) * NTOT + n2];
      }
    }
    vals[i] = v;
    ss += v * v;
  }
  #pragma unroll
  for (int m = 16; m >= 1; m >>= 1) ss += __shfl_xor(ss, m, 32);
  float inv = 1.0f / fmaxf(sqrtf(ss), 1e-4f);
  #pragma unroll
  for (int i = 0; i < 10; i++) W[lane + i * 32] = (_Float16)(vals[i] * inv);
}

// ---------------- kernel 4: raw filters transposed Rt [B][KRP][NP] (f16) ----------------
__global__ __launch_bounds__(256)
void rt_kernel(const float* __restrict__ base, _Float16* __restrict__ Rt)
{
  const int TOT = BATCH * KRP * NP;
  for (int idx = blockIdx.x * blockDim.x + threadIdx.x; idx < TOT;
       idx += gridDim.x * blockDim.x) {
    int n = idx % NP; int c9 = (idx / NP) % KRP; int b = idx / (NP * KRP);
    float v = 0.f;
    if (n < NTOT && c9 < KR) {
      int hs, p; decode_n(n, hs, p);
      int y = p / hs, x = p % hs;
      int c = c9 / 9, r9 = c9 % 9, dy = r9 / 3, dx = r9 % 3;
      int yy = y + dy - 1, xx = x + dx - 1;
      if (yy >= 0 && yy < hs && xx >= 0 && xx < hs) {
        int n2 = n + (dy - 1) * hs + (dx - 1);
        v = base[((size_t)b * CH + c) * NTOT + n2];
      }
    }
    Rt[idx] = (_Float16)v;
  }
}

// ---------------- WMMA fragment loaders (documented CDNA5 16-bit layouts) --------
// A 16x32: lanes 0-15 hold K {0..7,16..23}; lanes 16-31 hold K {8..15,24..31}
__device__ __forceinline__ h16 frag_a(const _Float16* p, int hi){
  union { h16 v; h8 h[2]; } u;
  u.h[0] = *(const h8*)(p + hi * 8);
  u.h[1] = *(const h8*)(p + 16 + hi * 8);
  return u.v;
}
// B 32x16: lanes 0-15 hold K {0..7,16..23} of column N=lane; lanes 16-31 hold K {8..15,24..31}
__device__ __forceinline__ h16 frag_b(const _Float16* p, int hi){
  union { h16 v; h8 h[2]; } u;
  u.h[0] = *(const h8*)(p + hi * 8);
  u.h[1] = *(const h8*)(p + 16 + hi * 8);
  return u.v;
}
// C/D 16x16 f32: VGPR i, lanes 0-15 -> M=i; lanes 16-31 -> M=8+i; N = lane&15
__device__ __forceinline__ void store_tile(float* C, int ldc, int row0, int col0,
                                           f8 acc, int r, int hi){
  float* p = C + (size_t)(row0 + hi * 8) * ldc + col0 + r;
  #pragma unroll
  for (int i = 0; i < 8; i++) p[(size_t)i * ldc] = acc[i];
}

// ---------------- async global -> LDS staging (CDNA5 ASYNCcnt path) ----------------
__device__ __forceinline__ void async_copy16(uint32_t lds_off, const _Float16* g){
  uint64_t ga = (uint64_t)(uintptr_t)g;
  asm volatile("global_load_async_to_lds_b128 %0, %1, off"
               :: "v"(lds_off), "v"(ga) : "memory");
}
__device__ __forceinline__ void wait_async0(){
  asm volatile("s_wait_asynccnt 0x0" ::: "memory");
}

// ---------------- kernels 5/7: LDS-staged batched GEMM ----------------
// C[b] (MxN) = A[b] (MxK, row-major) * B[b]^T (B stored NxK row-major)
// Block: 256 threads = 8 waves (2x4); block tile 128(M) x 256(N); K-step 64,
// double-buffered LDS. Wave tile 64x64 -> 4 A-frags x 4 B-frags per 32-chunk
// -> 32 WMMAs per barrier, 1:1 ds_load_b128 : WMMA.
#define BM 128
#define BN 256
#define KSTEP 64
#define LDS_K 72   // 64 data halves + 8 pad halves: rows advance 36 banks (all 16 distinct)

// copy ROWS x 64-half panel (row-major, leading dim ldg) into LDS stage
template<int ROWS>
__device__ __forceinline__ void stage_panel(const _Float16* g, int ldg,
                                            uint32_t lds_base, int tid){
  // ROWS rows * 128B = ROWS*8 chunks of 16B; 256 threads
  #pragma unroll
  for (int i = 0; i < ROWS / 32; i++) {
    int c   = tid + i * 256;
    int row = c >> 3;
    int col = (c & 7) * 8;                       // halves
    async_copy16(lds_base + (uint32_t)(row * LDS_K + col) * 2,
                 g + (size_t)row * ldg + col);
  }
}

__global__ __launch_bounds__(256)
void wmma_gemm_lds_kernel(const _Float16* __restrict__ A, const _Float16* __restrict__ Bm,
                          float* __restrict__ C, int M, int Nn, int K)
{
  __shared__ __align__(16) _Float16 As[2][BM * LDS_K];
  __shared__ __align__(16) _Float16 Bs[2][BN * LDS_K];

  int tid  = threadIdx.x;
  int wid  = tid >> 5, lane = tid & 31;
  int r    = lane & 15, hi = lane >> 4;

  int tn       = Nn / BN;
  int tilesPer = (M / BM) * tn;
  int blk = blockIdx.x;
  int b   = blk / tilesPer;
  int t   = blk - b * tilesPer;
  int im  = (t / tn) * BM;
  int in0 = (t % tn) * BN;

  const _Float16* Ab = A  + (size_t)b * M  * K + (size_t)im  * K;
  const _Float16* Bb = Bm + (size_t)b * Nn * K + (size_t)in0 * K;
  float*          Cb = C  + (size_t)b * M  * Nn;

  uint32_t aB[2] = { (uint32_t)(uintptr_t)&As[0][0], (uint32_t)(uintptr_t)&As[1][0] };
  uint32_t bB[2] = { (uint32_t)(uintptr_t)&Bs[0][0], (uint32_t)(uintptr_t)&Bs[1][0] };

  // wave grid 2x4 -> wave tile 64(M) x 64(N)
  int mo = (wid >> 2) * 64;
  int no = (wid & 3) * 64;

  f8 acc[4][4];
  #pragma unroll
  for (int i = 0; i < 4; i++)
    #pragma unroll
    for (int j = 0; j < 4; j++) acc[i][j] = (f8){};

  // prologue: fill stage 0
  stage_panel<BM>(Ab, K, aB[0], tid);
  stage_panel<BN>(Bb, K, bB[0], tid);
  wait_async0();
  __syncthreads();

  int nstage = K / KSTEP;
  for (int s = 0; s < nstage; s++) {
    int cur = s & 1, nxt = cur ^ 1;
    if (s + 1 < nstage) {                        // overlap next fill with compute
      stage_panel<BM>(Ab + (s + 1) * KSTEP, K, aB[nxt], tid);
      stage_panel<BN>(Bb + (s + 1) * KSTEP, K, bB[nxt], tid);
    }
    #pragma unroll
    for (int cc = 0; cc < 2; cc++) {             // two 32-K chunks per stage
      h16 af[4], bf[4];
      #pragma unroll
      for (int i = 0; i < 4; i++)
        af[i] = frag_a(&As[cur][(mo + i * 16 + r) * LDS_K + cc * 32], hi);
      #pragma unroll
      for (int j = 0; j < 4; j++)
        bf[j] = frag_b(&Bs[cur][(no + j * 16 + r) * LDS_K + cc * 32], hi);
      #pragma unroll
      for (int i = 0; i < 4; i++)
        #pragma unroll
        for (int j = 0; j < 4; j++)
          acc[i][j] = __builtin_amdgcn_wmma_f32_16x16x32_f16(
              false, af[i], false, bf[j], (short)0, acc[i][j], false, false);
    }
    wait_async0();                               // next stage resident
    __syncthreads();                             // everyone done reading cur
  }

  #pragma unroll
  for (int i = 0; i < 4; i++)
    #pragma unroll
    for (int j = 0; j < 4; j++)
      store_tile(Cb, Nn, im + mo + i * 16, in0 + no + j * 16, acc[i][j], r, hi);
}

// ---------------- kernel 6: softmax over N per (b, pixel); write Yt [B][HW][NP] f16 ----
// one block = one (batch, 32-pixel group): 8 waves stride N, LDS-combine partials
__global__ __launch_bounds__(256)
void softmax_kernel(const float* __restrict__ yi, _Float16* __restrict__ Yt)
{
  __shared__ float sm[8][32], sd[8][32];
  int tid = threadIdx.x; int wid = tid >> 5; int lane = tid & 31;
  int blk = blockIdx.x;                           // BATCH * (HW/32) blocks
  int b = blk >> 5;
  int p = ((blk & 31) << 5) + lane;
  const float* Y = yi + (size_t)b * NP * HW + p;
  float m = -1e30f, den = 0.f;
  for (int n = wid; n < NTOT; n += 8) {
    float v = Y[(size_t)n * HW] * 10.0f;          // softmax_scale
    float nm = fmaxf(m, v);
    den = den * __expf(m - nm) + __expf(v - nm);
    m = nm;
  }
  sm[wid][lane] = m; sd[wid][lane] = den;
  __syncthreads();
  float M = -1e30f;
  #pragma unroll
  for (int i = 0; i < 8; i++) M = fmaxf(M, sm[i][lane]);
  float D = 0.f;
  #pragma unroll
  for (int i = 0; i < 8; i++) D += sd[i][lane] * __expf(sm[i][lane] - M);
  float inv = 1.0f / D;
  _Float16* W = Yt + ((size_t)b * HW + p) * NP;
  for (int n = wid; n < NTOT; n += 8)
    W[n] = (_Float16)(__expf(Y[(size_t)n * HW] * 10.0f - M) * inv);
  for (int n = NTOT + wid; n < NP; n += 8)
    W[n] = (_Float16)0.f;                          // K-padding for GEMM2
}

// ---------------- kernel 8: fold (conv_transpose tap sum) /4 + residual ----------------
__global__ __launch_bounds__(256)
void fold_kernel(const float* __restrict__ z, const float* __restrict__ xin,
                 float* __restrict__ out)
{
  int idx = blockIdx.x * blockDim.x + threadIdx.x;
  if (idx >= BATCH * CH * HW) return;
  int p = idx % HW; int c = (idx / HW) % CH; int b = idx / (HW * CH);
  int y = p >> 5, x = p & 31;
  float s = 0.f;
  #pragma unroll
  for (int kh = 0; kh < 3; kh++) {
    int yy = y + 1 - kh;
    if (yy < 0 || yy >= 32) continue;
    #pragma unroll
    for (int kw = 0; kw < 3; kw++) {
      int xc = x + 1 - kw;
      if (xc < 0 || xc >= 32) continue;
      s += z[((size_t)b * KRP + c * 9 + kh * 3 + kw) * HW + yy * 32 + xc];
    }
  }
  out[idx] = s * 0.25f + xin[idx];
}

// ---------------- host launcher ----------------
extern "C" void kernel_launch(void* const* d_in, const int* in_sizes, int n_in,
                              void* d_out, int out_size, void* d_ws, size_t ws_size,
                              hipStream_t stream)
{
  const float* x      = (const float*)d_in[0];
  const float* w_base = (const float*)d_in[1];
  const float* b_base = (const float*)d_in[2];
  const float* a_base = (const float*)d_in[3];
  const float* w_match= (const float*)d_in[4];
  const float* b_match= (const float*)d_in[5];
  const float* a_match= (const float*)d_in[6];
  const float* w_asm  = (const float*)d_in[7];
  const float* b_asm  = (const float*)d_in[8];
  const float* a_asm  = (const float*)d_in[9];
  float* out = (float*)d_out;

  char* ws = (char*)d_ws;
  size_t off = 0;
  auto carve = [&](size_t bytes) -> char* {
    char* p = ws + off;
    off = (off + bytes + 255) & ~(size_t)255;
    return p;
  };
  float*    base_f = (float*)   carve((size_t)BATCH * CH  * NTOT * 4);
  float*    refm_f = (float*)   carve((size_t)BATCH * CH2 * NTOT * 4);
  float*    mb_f   = (float*)   carve((size_t)BATCH * CH2 * HW   * 4);
  _Float16* xp_h   = (_Float16*)carve((size_t)BATCH * HW  * KQP  * 2);
  _Float16* wi_h   = (_Float16*)carve((size_t)BATCH * NP  * KQP  * 2);
  _Float16* rt_h   = (_Float16*)carve((size_t)BATCH * KRP * NP   * 2);
  float*    yi_f   = (float*)   carve((size_t)BATCH * NP  * HW   * 4);
  _Float16* yt_h   = (_Float16*)carve((size_t)BATCH * HW  * NP   * 2);
  float*    z_f    = (float*)   carve((size_t)BATCH * KRP * HW   * 4);

  const int TPB = 256;

  // 1. activations (1x1 conv + PReLU, full-res + pyramid levels)
  {
    int total = BATCH * CH * NTOT + BATCH * CH2 * NTOT + BATCH * CH2 * HW;
    act_kernel<<<(total + TPB - 1) / TPB, TPB, 0, stream>>>(
        x, w_base, b_base, a_base, w_match, b_match, a_match,
        w_asm, b_asm, a_asm, base_f, refm_f, mb_f);
  }
  // 2. query patches (K zero-padded to KQP)
  {
    int total = BATCH * HW * KQP;
    xp_kernel<<<(total + TPB - 1) / TPB, TPB, 0, stream>>>(mb_f, xp_h);
  }
  // 3. normalized match filters
  {
    int waves = BATCH * NP;
    wi_kernel<<<(waves * 32 + TPB - 1) / TPB, TPB, 0, stream>>>(refm_f, wi_h);
  }
  // 4. raw filters (transposed, padded, for GEMM2 A)
  {
    int total = BATCH * KRP * NP;
    rt_kernel<<<(total + TPB - 1) / TPB, TPB, 0, stream>>>(base_f, rt_h);
  }
  // 5. GEMM1: yi[b] = wi[b] (NP x KQP) * xp[b]^T (HW x KQP)
  {
    int blocks = BATCH * (NP / BM) * (HW / BN);
    wmma_gemm_lds_kernel<<<blocks, TPB, 0, stream>>>(wi_h, xp_h, yi_f, NP, HW, KQP);
  }
  // 6. softmax over N, write transposed f16
  {
    int blocks = BATCH * (HW / 32);
    softmax_kernel<<<blocks, TPB, 0, stream>>>(yi_f, yt_h);
  }
  // 7. GEMM2: z[b] = Rt[b] (KRP x NP) * Yt[b]^T (HW x NP)
  {
    int blocks = BATCH * (KRP / BM) * (HW / BN);
    wmma_gemm_lds_kernel<<<blocks, TPB, 0, stream>>>(rt_h, yt_h, z_f, KRP, HW, NP);
  }
  // 8. fold + residual
  {
    int total = BATCH * CH * HW;
    fold_kernel<<<(total + TPB - 1) / TPB, TPB, 0, stream>>>(z_f, x, out);
  }
}
